// LSTM_21852793602349
// MI455X (gfx1250) — matrix operand order repaired and checked by reading, not compiled, not run on hardware
//
#include <hip/hip_runtime.h>

// ---------------------------------------------------------------------------
// 2-layer LSTM for MI455X (gfx1250, wave32, WMMA).
//   B=64, T=2048, D=H=150, 4H=600.
// K padded 150->160 (5 chunks of 32 for bf16 WMMA), N padded 600->608 (38 tiles).
// ---------------------------------------------------------------------------

typedef __attribute__((ext_vector_type(16))) __bf16 v16bf;
typedef __attribute__((ext_vector_type(8)))  float  v8f;

#define LSTM_H   150
#define LSTM_4H  600
#define LSTM_B   64
#define LSTM_T   2048
#define NKC      5                      // K chunks of 32 (K padded to 160)
#define NNT      38                     // N tiles of 16 (N padded to 608)
#define ZSTRIDE  612                    // zlds row stride (== 4 mod 8: conflict-free halves)
#define FRAG_EL  512                    // elements per 16x32 / 32x16 fragment
#define WPACK_ELEMS (NKC * NNT * FRAG_EL)   // 97280 bf16
#define WPACK_BYTES (WPACK_ELEMS * 2)       // 194560 B (fits: 320 KB LDS/WGP)
#define HPACK_ELEMS (NKC * FRAG_EL)         // h in A-fragment order: 2560 bf16

static __device__ __forceinline__ __bf16 f32_to_bf16(float f) {
    unsigned u = __builtin_bit_cast(unsigned, f);
    u = (u + 0x7FFFu + ((u >> 16) & 1u)) >> 16;         // round-to-nearest-even
    unsigned short s = (unsigned short)u;
    return __builtin_bit_cast(__bf16, s);
}

// Pack W [150 x 600] f32 row-major into LDS as bf16 WMMA B-fragments,
// lane-contiguous so each lane reads its 32B slice with 2x ds_load_b128.
// B-fragment layout (16-bit, 32x16): lane l -> column N = l&15,
// K = (e>>3)*16 + (l>>4)*8 + (e&7)  (e = 0..15).
static __device__ __forceinline__ void pack_W_to_lds(const float* __restrict__ W,
                                                     __bf16* wpack) {
    for (int i = threadIdx.x; i < WPACK_ELEMS; i += blockDim.x) {
        int e    = i & 15;
        int lane = (i >> 4) & 31;
        int kcnt = i >> 9;
        int kc   = kcnt / NNT;
        int nt   = kcnt - kc * NNT;
        int K    = kc * 32 + ((e >> 3) << 4) + ((lane >> 4) << 3) + (e & 7);
        int N    = (nt << 4) + (lane & 15);
        float v  = (K < LSTM_H && N < LSTM_4H) ? W[K * LSTM_4H + N] : 0.0f;
        wpack[i] = f32_to_bf16(v);
    }
}

// ---------------------------------------------------------------------------
// Kernel 1: input projection  out[M x 600] = in[M x 150] @ W[150 x 600]
// 8 waves / workgroup; each wave owns a 16-row M tile; 128 rows / workgroup.
// A-fragment layout (16-bit, 16x32): lane l -> row M = l&15,
// element e -> K = (e>>3)*16 + (l>>4)*8 + (e&7).
// ---------------------------------------------------------------------------
__global__ __launch_bounds__(256) void lstm_proj_wmma(
        const float* __restrict__ in,     // [M x 150]
        const float* __restrict__ Wx,     // [150 x 600]
        float* __restrict__ out)          // [M x 600]
{
    extern __shared__ __align__(16) char smem[];
    __bf16* wpack = (__bf16*)smem;
    pack_W_to_lds(Wx, wpack);
    __syncthreads();

    const int lane    = threadIdx.x & 31;
    const int wave    = threadIdx.x >> 5;
    const int m       = lane & 15;
    const int hl      = lane >> 4;
    const int rowBase = blockIdx.x * 128 + wave * 16;
    const float* arow = in + (size_t)(rowBase + m) * LSTM_H;

    v16bf afrag[NKC];
    #pragma unroll
    for (int kc = 0; kc < NKC; ++kc) {
        int k0 = kc * 32 + hl * 8;
        #pragma unroll
        for (int e = 0; e < 8; ++e) {
            int ka = k0 + e;
            int kb = k0 + 16 + e;
            afrag[kc][e]     = f32_to_bf16(ka < LSTM_H ? arow[ka] : 0.0f);
            afrag[kc][8 + e] = f32_to_bf16(kb < LSTM_H ? arow[kb] : 0.0f);
        }
    }

    for (int nt = 0; nt < NNT; ++nt) {
        v8f acc = {};
        #pragma unroll
        for (int kc = 0; kc < NKC; ++kc) {
            v16bf bfrag = *(const v16bf*)&wpack[((kc * NNT + nt) << 9) + (lane << 4)];
            acc = __builtin_amdgcn_wmma_f32_16x16x32_bf16(
                      false, afrag[kc], false, bfrag, (short)0, acc, false, false);
        }
        int col = (nt << 4) + m;
        if (col < LSTM_4H) {
            #pragma unroll
            for (int r = 0; r < 8; ++r) {
                int row = rowBase + hl * 8 + r;       // D layout: M = r + 8*(l>>4)
                out[(size_t)row * LSTM_4H + col] = acc[r];
            }
        }
    }
}

// ---------------------------------------------------------------------------
// Kernel 2: persistent recurrence. 4 workgroups x 16 batch rows, no cross-WG
// sync needed (recurrence is independent per batch row). Wh, h, c, z all
// resident in LDS across all 2048 steps.
// h is stored in LDS as bf16 *already in WMMA A-fragment order* (written by
// the gate phase), so the GEMM phase needs only 2x ds_load_b128 per fragment
// and zero conversion VALU before the first v_wmma issues.
// ---------------------------------------------------------------------------
__global__ __launch_bounds__(256) void lstm_recur_wmma(
        const float* __restrict__ xproj,  // [B*T x 600]
        const float* __restrict__ Wh,     // [150 x 600]
        const float* __restrict__ bias,   // [600]
        float* __restrict__ ys)           // [B*T x 150]
{
    extern __shared__ __align__(16) char smem[];
    __bf16* wpack = (__bf16*)smem;                       // 194560 B
    __bf16* hpack = (__bf16*)(smem + WPACK_BYTES);       // 2560 bf16 = 5120 B
    float*  clds  = (float*)(smem + WPACK_BYTES + HPACK_ELEMS * 2);  // [16][150]
    float*  zlds  = clds + 16 * LSTM_H;                  // [16][ZSTRIDE]

    pack_W_to_lds(Wh, wpack);
    for (int i = threadIdx.x; i < HPACK_ELEMS;  i += blockDim.x) hpack[i] = (__bf16)0.0f;
    for (int i = threadIdx.x; i < 16 * LSTM_H;  i += blockDim.x) clds[i] = 0.0f;
    __syncthreads();

    const int lane = threadIdx.x & 31;
    const int wave = threadIdx.x >> 5;
    const int m    = lane & 15;
    const int hl   = lane >> 4;
    const int rb   = blockIdx.x * 16;                    // 16 batch rows per WG

    for (int t = 0; t < LSTM_T; ++t) {
        // --- recurrent GEMM: z[16x600] = h[16x160] @ Wh[160x608] ------------
        v16bf afrag[NKC];
        #pragma unroll
        for (int kc = 0; kc < NKC; ++kc)
            afrag[kc] = *(const v16bf*)&hpack[(kc << 9) + (lane << 4)];

        for (int nt = wave; nt < NNT; nt += 8) {
            v8f acc = {};
            #pragma unroll
            for (int kc = 0; kc < NKC; ++kc) {
                v16bf bfrag = *(const v16bf*)&wpack[((kc * NNT + nt) << 9) + (lane << 4)];
                acc = __builtin_amdgcn_wmma_f32_16x16x32_bf16(
                          false, afrag[kc], false, bfrag, (short)0, acc, false, false);
            }
            #pragma unroll
            for (int r = 0; r < 8; ++r)
                zlds[(hl * 8 + r) * ZSTRIDE + (nt << 4) + m] = acc[r];
        }
        __syncthreads();

        // --- gates + state update (f32) -------------------------------------
        for (int idx = threadIdx.x; idx < 16 * LSTM_H; idx += blockDim.x) {
            int row = idx / LSTM_H;
            int col = idx - row * LSTM_H;
            size_t gbase = ((size_t)(rb + row) * LSTM_T + t) * LSTM_4H + col;
            float zi = zlds[row * ZSTRIDE + col      ] + xproj[gbase      ] + bias[col      ];
            float zf = zlds[row * ZSTRIDE + col + 150] + xproj[gbase + 150] + bias[col + 150];
            float zg = zlds[row * ZSTRIDE + col + 300] + xproj[gbase + 300] + bias[col + 300];
            float zo = zlds[row * ZSTRIDE + col + 450] + xproj[gbase + 450] + bias[col + 450];
            float ig = 1.0f / (1.0f + __expf(-zi));
            float fg = 1.0f / (1.0f + __expf(-zf));
            float gg = tanhf(zg);
            float og = 1.0f / (1.0f + __expf(-zo));
            float c  = fg * clds[idx] + ig * gg;
            clds[idx] = c;
            float h  = og * tanhf(c);
            // scatter h into A-fragment order:
            //   k' = col%32; lane = row + 16*((k'>>3)&1); e = (k'>>4)*8 + (k'&7)
            int kp = col & 31;
            int dl = row + (((kp >> 3) & 1) << 4);
            int de = ((kp >> 4) << 3) + (kp & 7);
            hpack[((col >> 5) << 9) + (dl << 4) + de] = f32_to_bf16(h);
            ys[((size_t)(rb + row) * LSTM_T + t) * LSTM_H + col] = h;
            if (t + 1 < LSTM_T) {   // warm all four gate segments for t+1
                __builtin_prefetch(&xproj[gbase + LSTM_4H      ], 0, 1);
                __builtin_prefetch(&xproj[gbase + LSTM_4H + 150], 0, 1);
                __builtin_prefetch(&xproj[gbase + LSTM_4H + 300], 0, 1);
                __builtin_prefetch(&xproj[gbase + LSTM_4H + 450], 0, 1);
            }
        }
        __syncthreads();
    }
}

// ---------------------------------------------------------------------------
extern "C" void kernel_launch(void* const* d_in, const int* in_sizes, int n_in,
                              void* d_out, int out_size, void* d_ws, size_t ws_size,
                              hipStream_t stream) {
    (void)in_sizes; (void)n_in; (void)out_size; (void)ws_size;
    const float* xs  = (const float*)d_in[0];
    const float* Wx0 = (const float*)d_in[1];
    const float* Wh0 = (const float*)d_in[2];
    const float* b0  = (const float*)d_in[3];
    const float* Wx1 = (const float*)d_in[4];
    const float* Wh1 = (const float*)d_in[5];
    const float* b1  = (const float*)d_in[6];
    float* out = (float*)d_out;

    const size_t M = (size_t)LSTM_B * LSTM_T;            // 131072 rows
    float* proj = (float*)d_ws;                          // M*600 f32 (~315 MB)
    float* ys0  = proj + M * LSTM_4H;                    // M*150 f32 (~79 MB)

    const size_t projLds  = WPACK_BYTES;
    const size_t recurLds = WPACK_BYTES + (size_t)HPACK_ELEMS * 2 +
                            (size_t)(16 * LSTM_H + 16 * ZSTRIDE) * sizeof(float);

    dim3 blk(256);
    // Layer 0
    lstm_proj_wmma <<<dim3((unsigned)(M / 128)), blk, projLds,  stream>>>(xs,  Wx0, proj);
    lstm_recur_wmma<<<dim3(LSTM_B / 16),          blk, recurLds, stream>>>(proj, Wh0, b0, ys0);
    // Layer 1
    lstm_proj_wmma <<<dim3((unsigned)(M / 128)), blk, projLds,  stream>>>(ys0, Wx1, proj);
    lstm_recur_wmma<<<dim3(LSTM_B / 16),          blk, recurLds, stream>>>(proj, Wh1, b1, out);
}
